// PaiNN_22600117912023
// MI455X (gfx1250) — matrix operand order repaired and checked by reading, not compiled, not run on hardware
//
#include <hip/hip_runtime.h>
#include <hip/hip_bf16.h>

#define FDIM 128
#define NRBF 20

typedef __attribute__((ext_vector_type(16))) _Float16 v16h;
typedef __attribute__((ext_vector_type(8)))  float    v8f;

__device__ __forceinline__ float silu_f(float x) { return x / (1.0f + expf(-x)); }

// ---------------- conversion kernels ----------------
__global__ void pk_cvt(const float* __restrict__ a, _Float16* __restrict__ o, size_t n) {
  size_t i = (size_t)blockIdx.x * blockDim.x + threadIdx.x;
  size_t st = (size_t)gridDim.x * blockDim.x;
  for (; i < n; i += st) o[i] = (_Float16)a[i];
}
__global__ void pk_cvt_silu(const float* __restrict__ a, _Float16* __restrict__ o, size_t n) {
  size_t i = (size_t)blockIdx.x * blockDim.x + threadIdx.x;
  size_t st = (size_t)gridDim.x * blockDim.x;
  for (; i < n; i += st) o[i] = (_Float16)silu_f(a[i]);
}

// ---------------- generic WMMA GEMM (32x32 macro-tile per wave) ----------------
// C[M,N] = act( A[M,K] (f16, optionally row-gathered) @ W[N,K]^T (f16) + bias )
// Each wave computes a 32x32 output block as four 16x16 WMMA tiles:
// 2 A-frags + 2 B-frags per k-step feed 4 independent v_wmma_f32_16x16x32_f16,
// doubling FLOP/byte vs one-tile-per-wave and removing WMMA->WMMA RAW hazards.
// Fragment layouts per CDNA5 ISA 7.12.2:
//  A 16x32 f16 : lanes 0-15 -> M=lane, halves = K {0..7,16..23}; lanes 16-31 -> K+8
//  B 32x16 f16 : lane&15 -> N, halves = K {0..7,16..23}; lanes 16-31 -> K+8
//  C/D 16x16 f32: vgpr r -> M = r + 8*(lane>=16), N = lane&15
__device__ __forceinline__ void pk_store_tile(const v8f& acc, int mBase, int nBase,
                                              int laneLo, int laneHi,
                                              const float* __restrict__ bias,
                                              float* __restrict__ Cf,
                                              _Float16* __restrict__ Ch,
                                              int ldc, int M, int N, int act) {
  int n = nBase + laneLo;
  if (n >= N) return;
  float bz = bias ? bias[n] : 0.0f;
#pragma unroll
  for (int r = 0; r < 8; ++r) {
    int m = mBase + r + laneHi * 8;
    if (m < M) {
      float x = acc[r] + bz;
      if (act == 1) x = silu_f(x);
      else if (act == 2) x = silu_f(x) * (1.0f / 0.6f);
      long long off = (long long)m * ldc + n;
      if (Cf) Cf[off] = x;
      if (Ch) Ch[off] = (_Float16)x;
    }
  }
}

__global__ void pk_gemm(const _Float16* __restrict__ A, int lda,
                        const int* __restrict__ rowIdx,
                        const _Float16* __restrict__ W, int ldw,
                        const float* __restrict__ bias,
                        float* __restrict__ Cf, _Float16* __restrict__ Ch, int ldc,
                        int M, int N, int K, int act) {
  int lane   = threadIdx.x & 31;
  long long wave = (long long)blockIdx.x * (blockDim.x >> 5) + (threadIdx.x >> 5);
  int tilesN = (N + 31) >> 5;
  long long tiles = (long long)((M + 31) >> 5) * tilesN;
  if (wave >= tiles) return;
  int tm = (int)(wave / tilesN), tn = (int)(wave % tilesN);
  int mBase = tm << 5, nBase = tn << 5;
  int laneLo = lane & 15, laneHi = lane >> 4;

  int ar0 = mBase + laneLo;      if (ar0 >= M) ar0 = M - 1;
  int ar1 = mBase + 16 + laneLo; if (ar1 >= M) ar1 = M - 1;
  long long arow0 = rowIdx ? (long long)rowIdx[ar0] : (long long)ar0;
  long long arow1 = rowIdx ? (long long)rowIdx[ar1] : (long long)ar1;
  const _Float16* Ap0 = A + arow0 * (long long)lda + laneHi * 8;
  const _Float16* Ap1 = A + arow1 * (long long)lda + laneHi * 8;

  int bn0 = nBase + laneLo;      if (bn0 >= N) bn0 = N - 1;
  int bn1 = nBase + 16 + laneLo; if (bn1 >= N) bn1 = N - 1;
  const _Float16* Wp0 = W + (long long)bn0 * ldw + laneHi * 8;
  const _Float16* Wp1 = W + (long long)bn1 * ldw + laneHi * 8;

  v8f acc00 = {}, acc01 = {}, acc10 = {}, acc11 = {};
  for (int kb = 0; kb < K; kb += 32) {
    v16h a0, a1, b0, b1;
#pragma unroll
    for (int e = 0; e < 8; ++e) {
      a0[e] = Ap0[kb + e]; a0[e + 8] = Ap0[kb + 16 + e];
      a1[e] = Ap1[kb + e]; a1[e + 8] = Ap1[kb + 16 + e];
    }
#pragma unroll
    for (int e = 0; e < 8; ++e) {
      b0[e] = Wp0[kb + e]; b0[e + 8] = Wp0[kb + 16 + e];
      b1[e] = Wp1[kb + e]; b1[e + 8] = Wp1[kb + 16 + e];
    }
    acc00 = __builtin_amdgcn_wmma_f32_16x16x32_f16(false, a0, false, b0, (short)0, acc00, false, false);
    acc01 = __builtin_amdgcn_wmma_f32_16x16x32_f16(false, a0, false, b1, (short)0, acc01, false, false);
    acc10 = __builtin_amdgcn_wmma_f32_16x16x32_f16(false, a1, false, b0, (short)0, acc10, false, false);
    acc11 = __builtin_amdgcn_wmma_f32_16x16x32_f16(false, a1, false, b1, (short)0, acc11, false, false);
  }
  pk_store_tile(acc00, mBase,      nBase,      laneLo, laneHi, bias, Cf, Ch, ldc, M, N, act);
  pk_store_tile(acc01, mBase,      nBase + 16, laneLo, laneHi, bias, Cf, Ch, ldc, M, N, act);
  pk_store_tile(acc10, mBase + 16, nBase,      laneLo, laneHi, bias, Cf, Ch, ldc, M, N, act);
  pk_store_tile(acc11, mBase + 16, nBase + 16, laneLo, laneHi, bias, Cf, Ch, ldc, M, N, act);
}

// ---------------- edge geometry ----------------
__global__ void pk_edge_prep(const float* __restrict__ pos, const int* __restrict__ ei,
                             int E, float* __restrict__ rbf, float* __restrict__ cutb,
                             float* __restrict__ nrm) {
  int e = blockIdx.x * blockDim.x + threadIdx.x;
  if (e >= E) return;
  int j = ei[e], i = ei[E + e];
  float dx = pos[j * 3 + 0] - pos[i * 3 + 0];
  float dy = pos[j * 3 + 1] - pos[i * 3 + 1];
  float dz = pos[j * 3 + 2] - pos[i * 3 + 2];
  float d = sqrtf(dx * dx + dy * dy + dz * dz);
  float safe = (d == 0.0f) ? 1.0f : d;
  const float kf = 3.14159265358979323846f / 5.0f;
#pragma unroll
  for (int r = 0; r < NRBF; ++r)
    rbf[(size_t)e * NRBF + r] = sinf(d * kf * (float)(r + 1)) / safe;
  cutb[e] = 0.5f * (cosf(d * kf) + 1.0f) * ((d < 5.0f) ? 1.0f : 0.0f);
  float inv = 1.0f / fmaxf(d, 1e-12f);
  nrm[(size_t)e * 3 + 0] = dx * inv;
  nrm[(size_t)e * 3 + 1] = dy * inv;
  nrm[(size_t)e * 3 + 2] = dz * inv;
}

// ---------------- edge message + scatter (Wf recomputed on the fly) ----------------
__global__ void pk_scatter(const int* __restrict__ ei, int E,
                           const float* __restrict__ rbf, const float* __restrict__ cutb,
                           const float* __restrict__ nrm,
                           const _Float16* __restrict__ phih,
                           const float* __restrict__ rbfW, const float* __restrict__ rbfb,
                           const float* __restrict__ v, float* __restrict__ dsacc,
                           float* __restrict__ dvacc, size_t PS) {
  int e = blockIdx.x;
  int f = threadIdx.x;  // 0..127
  __shared__ float rs[NRBF];
  __shared__ float sc;
  __shared__ float sn[3];
  __shared__ int si, sj;
  if (f < NRBF) rs[f] = rbf[(size_t)e * NRBF + f];
  if (f == 0) { sc = cutb[e]; si = ei[E + e]; sj = ei[e]; }
  if (f < 3) sn[f] = nrm[(size_t)e * 3 + f];
  __syncthreads();
  float wl = rbfb[f], wm = rbfb[f + 128], wr = rbfb[f + 256];
#pragma unroll
  for (int r = 0; r < NRBF; ++r) {
    float rv = rs[r];
    wl += rv * rbfW[(size_t)f * NRBF + r];
    wm += rv * rbfW[(size_t)(f + 128) * NRBF + r];
    wr += rv * rbfW[(size_t)(f + 256) * NRBF + r];
  }
  float c = sc;
  wl *= c; wm *= c; wr *= c;
  size_t pb = (size_t)e * 384;
  float left  = (float)phih[pb + f] * wl;
  float dsm   = (float)phih[pb + 128 + f] * wm;
  float right = (float)phih[pb + 256 + f] * wr;
  size_t ni = (size_t)si * FDIM + f, nj = (size_t)sj * FDIM + f;
  atomicAdd(&dsacc[ni], dsm);
#pragma unroll
  for (int cc = 0; cc < 3; ++cc) {
    float vj = v[(size_t)cc * PS + nj];
    atomicAdd(&dvacc[(size_t)cc * PS + ni], vj * left + right * sn[cc]);
  }
}

// ---------------- s/v accumulate update ----------------
__global__ void pk_apply(float* __restrict__ s, _Float16* __restrict__ sh,
                         float* __restrict__ v, _Float16* __restrict__ vh,
                         const float* __restrict__ dsacc, const float* __restrict__ dvacc,
                         size_t PS) {
  size_t i = (size_t)blockIdx.x * blockDim.x + threadIdx.x;
  size_t st = (size_t)gridDim.x * blockDim.x;
  for (; i < PS; i += st) {
    float sv = s[i] + dsacc[i];
    s[i] = sv; sh[i] = (_Float16)sv;
#pragma unroll
    for (int c = 0; c < 3; ++c) {
      float vv = v[c * PS + i] + dvacc[c * PS + i];
      v[c * PS + i] = vv; vh[c * PS + i] = (_Float16)vv;
    }
  }
}

__global__ void pk_uv_post(const float* __restrict__ U, const float* __restrict__ V,
                           size_t PS, float* __restrict__ UV, float* __restrict__ nV) {
  size_t i = (size_t)blockIdx.x * blockDim.x + threadIdx.x;
  size_t st = (size_t)gridDim.x * blockDim.x;
  for (; i < PS; i += st) {
    float uv = 0.f, vv = 0.f;
#pragma unroll
    for (int c = 0; c < 3; ++c) {
      float u = U[c * PS + i], w = V[c * PS + i];
      uv += u * w; vv += w * w;
    }
    UV[i] = uv; nV[i] = sqrtf(vv);
  }
}

__global__ void pk_cat2(const float* __restrict__ a, const float* __restrict__ b,
                        _Float16* __restrict__ o, int n, int wa, int wb) {
  int w = wa + wb;
  size_t tot = (size_t)n * w;
  size_t i = (size_t)blockIdx.x * blockDim.x + threadIdx.x;
  size_t st = (size_t)gridDim.x * blockDim.x;
  for (; i < tot; i += st) {
    size_t row = i / w; int col = (int)(i % w);
    float x = (col < wa) ? a[row * wa + col] : b[row * wb + (col - wa)];
    o[i] = (_Float16)x;
  }
}

__global__ void pk_layer_update(float* __restrict__ s, _Float16* __restrict__ sh,
                                float* __restrict__ v, _Float16* __restrict__ vh,
                                const float* __restrict__ su, const float* __restrict__ UV,
                                size_t PS) {
  size_t i = (size_t)blockIdx.x * blockDim.x + threadIdx.x;
  size_t st = (size_t)gridDim.x * blockDim.x;
  for (; i < PS; i += st) {
    size_t n = i >> 7; int f = (int)(i & 127);
    size_t b = n * 384 + f;
    float top = su[b], mid = su[b + 128], bot = su[b + 256];
    float sv = s[i] + mid * UV[i] + bot;
    s[i] = sv; sh[i] = (_Float16)sv;
#pragma unroll
    for (int c = 0; c < 3; ++c) {
      float vv = v[c * PS + i] * (1.0f + top);
      v[c * PS + i] = vv; vh[c * PS + i] = (_Float16)vv;
    }
  }
}

__global__ void pk_norm3(const float* __restrict__ w, size_t plane,
                         float* __restrict__ out, size_t n) {
  size_t i = (size_t)blockIdx.x * blockDim.x + threadIdx.x;
  size_t st = (size_t)gridDim.x * blockDim.x;
  for (; i < n; i += st) {
    float a = w[i], b = w[plane + i], c = w[2 * plane + i];
    out[i] = sqrtf(a * a + b * b + c * c);
  }
}

__global__ void pk_blk0_post(const float* __restrict__ h2, const float* __restrict__ vec2,
                             size_t P64, float* __restrict__ xb, _Float16* __restrict__ xbh,
                             _Float16* __restrict__ vecbh, size_t n64) {
  size_t i = (size_t)blockIdx.x * blockDim.x + threadIdx.x;
  size_t st = (size_t)gridDim.x * blockDim.x;
  for (; i < n64; i += st) {
    size_t n = i >> 6; int f = (int)(i & 63);
    float xv = silu_f(h2[n * 128 + f]);
    xb[i] = xv; xbh[i] = (_Float16)xv;
    float g = h2[n * 128 + 64 + f];
#pragma unroll
    for (int c = 0; c < 3; ++c)
      vecbh[c * P64 + i] = (_Float16)(g * vec2[c * P64 + i]);
  }
}

__global__ void pk_vec2b(const _Float16* __restrict__ vecbh, size_t P64,
                         const float* __restrict__ w, float* __restrict__ out, int Nn) {
  int tot = Nn * 3;
  int i = blockIdx.x * blockDim.x + threadIdx.x;
  int st = gridDim.x * blockDim.x;
  for (; i < tot; i += st) {
    int n = i / 3, c = i % 3;
    float acc = 0.f;
#pragma unroll
    for (int g = 0; g < 64; ++g)
      acc += (float)vecbh[c * P64 + (size_t)n * 64 + g] * w[g];
    out[i] = acc;
  }
}

__global__ void pk_final(const float* __restrict__ hb, const float* __restrict__ w2,
                         const float* __restrict__ b2, const float* __restrict__ vec2b,
                         const float* __restrict__ pos, float* __restrict__ outv, int Nn) {
  int n = blockIdx.x * blockDim.x + threadIdx.x;
  int st = gridDim.x * blockDim.x;
  for (; n < Nn; n += st) {
    float g = b2[1];
#pragma unroll
    for (int k = 0; k < 64; ++k) g += hb[(size_t)n * 64 + k] * w2[64 + k];
#pragma unroll
    for (int c = 0; c < 3; ++c)
      outv[n * 3 + c] = pos[n * 3 + c] + g * vec2b[n * 3 + c];
  }
}

// ---------------- host side ----------------
static inline int grid1(size_t n) {
  size_t b = (n + 255) / 256;
  if (b > 16384) b = 16384;
  return (int)b;
}

static void launch_gemm(hipStream_t st, const _Float16* A, int lda, const int* rowIdx,
                        const _Float16* W, int ldw, const float* bias,
                        float* Cf, _Float16* Ch, int ldc, int M, int N, int K, int act) {
  long long tiles = (long long)((M + 31) / 32) * ((N + 31) / 32);
  int blocks = (int)((tiles + 7) / 8);
  pk_gemm<<<blocks, 256, 0, st>>>(A, lda, rowIdx, W, ldw, bias, Cf, Ch, ldc, M, N, K, act);
}

extern "C" void kernel_launch(void* const* d_in, const int* in_sizes, int n_in,
                              void* d_out, int out_size, void* d_ws, size_t ws_size,
                              hipStream_t stream) {
  const int Nn = in_sizes[1] / 3;      // 10000
  const int E  = in_sizes[2] / 2;      // 160000
  const size_t PS  = (size_t)Nn * FDIM;
  const size_t P64 = (size_t)Nn * 64;

  // ---- input pointers (pytree order: sorted dict keys inside params) ----
  const float* s_in = (const float*)d_in[0];
  const float* pos  = (const float*)d_in[1];
  const int*   ei   = (const int*)d_in[2];
  const float* embW = (const float*)d_in[3];
  const float* embB = (const float*)d_in[4];
  // layers: base 5 + 12*l : U_W,V_W,m1_W,m1_b,m2_W,m2_b,rbf_W,rbf_b,u2_W,u2_b,up_W,up_b
  const float* linW = (const float*)d_in[41];
  const float* linB = (const float*)d_in[42];
  // out_blocks: b0 @43: n0_W,n0_b,n2_W,n2_b,v1_W,v2_W ; b1 @49 same
  const float* n0W0 = (const float*)d_in[43]; const float* n0B0 = (const float*)d_in[44];
  const float* n2W0 = (const float*)d_in[45]; const float* n2B0 = (const float*)d_in[46];
  const float* v1W0 = (const float*)d_in[47]; const float* v2W0 = (const float*)d_in[48];
  const float* n0W1 = (const float*)d_in[49]; const float* n0B1 = (const float*)d_in[50];
  const float* n2W1 = (const float*)d_in[51]; const float* n2B1 = (const float*)d_in[52];
  const float* v1W1 = (const float*)d_in[53]; const float* v2W1 = (const float*)d_in[54];
  (void)n_in; (void)out_size; (void)ws_size;

  // ---- workspace bump allocator ----
  char* wp = (char*)d_ws;
  auto alloc = [&](size_t bytes) -> void* {
    void* r = (void*)wp;
    wp += (bytes + 255) & ~(size_t)255;
    return r;
  };
  float*    rbf    = (float*)alloc((size_t)E * NRBF * 4);
  float*    cutb   = (float*)alloc((size_t)E * 4);
  float*    nrm    = (float*)alloc((size_t)E * 3 * 4);
  float*    sbuf   = (float*)alloc(PS * 4);
  _Float16* sh     = (_Float16*)alloc(PS * 2);
  _Float16* s_in_h = (_Float16*)alloc(PS * 2);
  float*    vbuf   = (float*)alloc(3 * PS * 4);
  _Float16* vh     = (_Float16*)alloc(3 * PS * 2);
  _Float16* phi1h  = (_Float16*)alloc((size_t)E * 128 * 2);
  _Float16* phih   = (_Float16*)alloc((size_t)E * 384 * 2);
  float*    dsacc  = (float*)alloc(PS * 4);
  float*    dvacc  = (float*)alloc(3 * PS * 4);
  float*    Ub     = (float*)alloc(3 * PS * 4);
  float*    Vb     = (float*)alloc(3 * PS * 4);
  float*    UVb    = (float*)alloc(PS * 4);
  float*    nVb    = (float*)alloc(PS * 4);
  _Float16* cath   = (_Float16*)alloc((size_t)Nn * 256 * 2);
  float*    t1     = (float*)alloc(PS * 4);
  _Float16* t1h    = (_Float16*)alloc(PS * 2);
  float*    su     = (float*)alloc((size_t)Nn * 384 * 4);
  _Float16* cat2h  = (_Float16*)alloc((size_t)Nn * 256 * 2);
  _Float16* hh     = (_Float16*)alloc(PS * 2);
  float*    vec2   = (float*)alloc(3 * P64 * 4);
  float*    xb     = (float*)alloc(P64 * 4);
  _Float16* xbh    = (_Float16*)alloc(P64 * 2);
  _Float16* vecbh  = (_Float16*)alloc(3 * P64 * 2);
  _Float16* catbh  = (_Float16*)alloc((size_t)Nn * 128 * 2);
  float*    vec2b  = (float*)alloc((size_t)Nn * 3 * 4);
  _Float16* ssilh  = (_Float16*)alloc(PS * 2);
  // reuses: w1=Ub, vec1=UVb, h2=t1, w1b=Vb, vec1b=nVb, hb=dsacc
  float* w1 = Ub; float* vec1 = UVb; float* h2 = t1;
  float* w1b = Vb; float* vec1b = nVb; float* hb = dsacc;

  // f16 weight copies
  _Float16* embWh = (_Float16*)alloc(16384 * 2);
  _Float16* linWh = (_Float16*)alloc(16384 * 2);
  _Float16 *UWh[3], *VWh[3], *m1Wh[3], *m2Wh[3], *u2Wh[3], *upWh[3];
  for (int l = 0; l < 3; ++l) {
    UWh[l]  = (_Float16*)alloc(16384 * 2);
    VWh[l]  = (_Float16*)alloc(16384 * 2);
    m1Wh[l] = (_Float16*)alloc(16384 * 2);
    m2Wh[l] = (_Float16*)alloc(49152 * 2);
    u2Wh[l] = (_Float16*)alloc(49152 * 2);
    upWh[l] = (_Float16*)alloc(32768 * 2);
  }
  _Float16* n0W0h = (_Float16*)alloc(32768 * 2);
  _Float16* n2W0h = (_Float16*)alloc(16384 * 2);
  _Float16* v1W0h = (_Float16*)alloc(16384 * 2);
  _Float16* v2W0h = (_Float16*)alloc(8192 * 2);
  _Float16* n0W1h = (_Float16*)alloc(8192 * 2);
  _Float16* v1W1h = (_Float16*)alloc(4096 * 2);

#define CV(src, dst, n) pk_cvt<<<grid1(n), 256, 0, stream>>>((src), (dst), (size_t)(n))
  CV(embW, embWh, 16384);
  CV(linW, linWh, 16384);
  for (int l = 0; l < 3; ++l) {
    const float* const* L = (const float* const*)(d_in + 5 + 12 * l);
    CV(L[0], UWh[l], 16384);
    CV(L[1], VWh[l], 16384);
    CV(L[2], m1Wh[l], 16384);
    CV(L[4], m2Wh[l], 49152);
    CV(L[8], u2Wh[l], 49152);
    CV(L[10], upWh[l], 32768);
  }
  CV(n0W0, n0W0h, 32768);
  CV(n2W0, n2W0h, 16384);
  CV(v1W0, v1W0h, 16384);
  CV(v2W0, v2W0h, 8192);
  CV(n0W1, n0W1h, 8192);
  CV(v1W1, v1W1h, 4096);

  // ---- edge geometry ----
  pk_edge_prep<<<(E + 255) / 256, 256, 0, stream>>>(pos, ei, E, rbf, cutb, nrm);

  // ---- embedding: s = s_in @ embW^T + b ----
  CV(s_in, s_in_h, PS);
  launch_gemm(stream, s_in_h, 128, nullptr, embWh, 128, embB, sbuf, sh, 128,
              Nn, 128, 128, 0);

  // ---- v = 0 ----
  hipMemsetAsync(vbuf, 0, 3 * PS * 4, stream);
  hipMemsetAsync(vh, 0, 3 * PS * 2, stream);

  // ---- interaction layers ----
  for (int l = 0; l < 3; ++l) {
    const float* const* L = (const float* const*)(d_in + 5 + 12 * l);
    const float* m1B = L[3]; const float* m2B = L[5];
    const float* rbfW = L[6]; const float* rbfB = L[7];
    const float* u2B = L[9]; const float* upB = L[11];

    // phi1 = silu(s[j] @ m1^T + b) ; phi = phi1 @ m2^T + b   (edge-level WMMA GEMMs)
    launch_gemm(stream, sh, 128, ei, m1Wh[l], 128, m1B, nullptr, phi1h, 128,
                E, 128, 128, 1);
    launch_gemm(stream, phi1h, 128, nullptr, m2Wh[l], 128, m2B, nullptr, phih, 384,
                E, 384, 128, 0);

    // scatter messages (Wf recomputed from rbf on the fly)
    hipMemsetAsync(dsacc, 0, PS * 4, stream);
    hipMemsetAsync(dvacc, 0, 3 * PS * 4, stream);
    pk_scatter<<<E, 128, 0, stream>>>(ei, E, rbf, cutb, nrm, phih, rbfW, rbfB,
                                      vbuf, dsacc, dvacc, PS);
    pk_apply<<<grid1(PS), 256, 0, stream>>>(sbuf, sh, vbuf, vh, dsacc, dvacc, PS);

    // U,V per coordinate
    for (int c = 0; c < 3; ++c) {
      launch_gemm(stream, vh + c * PS, 128, nullptr, UWh[l], 128, nullptr,
                  Ub + c * PS, nullptr, 128, Nn, 128, 128, 0);
      launch_gemm(stream, vh + c * PS, 128, nullptr, VWh[l], 128, nullptr,
                  Vb + c * PS, nullptr, 128, Nn, 128, 128, 0);
    }
    pk_uv_post<<<grid1(PS), 256, 0, stream>>>(Ub, Vb, PS, UVb, nVb);
    pk_cat2<<<grid1((size_t)Nn * 256), 256, 0, stream>>>(sbuf, nVb, cath, Nn, 128, 128);

    // su = u2( silu( up( cat ) ) )
    launch_gemm(stream, cath, 256, nullptr, upWh[l], 256, upB, t1, t1h, 128,
                Nn, 128, 256, 1);
    launch_gemm(stream, t1h, 128, nullptr, u2Wh[l], 128, u2B, su, nullptr, 384,
                Nn, 384, 128, 0);
    pk_layer_update<<<grid1(PS), 256, 0, stream>>>(sbuf, sh, vbuf, vh, su, UVb, PS);
  }

  // ---- output block 0 (hid=128 -> 64) ----
  for (int c = 0; c < 3; ++c) {
    launch_gemm(stream, vh + c * PS, 128, nullptr, v1W0h, 128, nullptr,
                w1 + c * PS, nullptr, 128, Nn, 128, 128, 0);
    launch_gemm(stream, vh + c * PS, 128, nullptr, v2W0h, 128, nullptr,
                vec2 + c * P64, nullptr, 64, Nn, 64, 128, 0);
  }
  pk_norm3<<<grid1(PS), 256, 0, stream>>>(w1, PS, vec1, PS);
  pk_cat2<<<grid1((size_t)Nn * 256), 256, 0, stream>>>(sbuf, vec1, cat2h, Nn, 128, 128);
  launch_gemm(stream, cat2h, 256, nullptr, n0W0h, 256, n0B0, nullptr, hh, 128,
              Nn, 128, 256, 2);
  launch_gemm(stream, hh, 128, nullptr, n2W0h, 128, n2B0, h2, nullptr, 128,
              Nn, 128, 128, 0);
  pk_blk0_post<<<grid1(P64), 256, 0, stream>>>(h2, vec2, P64, xb, xbh, vecbh, P64);

  // ---- output block 1 (hid=64 -> 1) ----
  for (int c = 0; c < 3; ++c)
    launch_gemm(stream, vecbh + c * P64, 64, nullptr, v1W1h, 64, nullptr,
                w1b + c * P64, nullptr, 64, Nn, 64, 64, 0);
  pk_norm3<<<grid1(P64), 256, 0, stream>>>(w1b, P64, vec1b, P64);
  pk_cat2<<<grid1((size_t)Nn * 128), 256, 0, stream>>>(xb, vec1b, catbh, Nn, 64, 64);
  launch_gemm(stream, catbh, 128, nullptr, n0W1h, 128, n0B1, hb, nullptr, 64,
              Nn, 64, 128, 2);
  pk_vec2b<<<grid1((size_t)Nn * 3), 256, 0, stream>>>(vecbh, P64, v2W1, vec2b, Nn);
  float* outv = (float*)d_out + PS;  // v_out after s_out
  pk_final<<<grid1((size_t)Nn), 256, 0, stream>>>(hb, n2W1, n2B1, vec2b, pos, outv, Nn);

  // ---- s_out = silu(s) @ linW^T + b ----
  pk_cvt_silu<<<grid1(PS), 256, 0, stream>>>(sbuf, ssilh, PS);
  launch_gemm(stream, ssilh, 128, nullptr, linWh, 128, linB, (float*)d_out, nullptr, 128,
              Nn, 128, 128, 0);
}
